// CEN_53936199303805
// MI455X (gfx1250) — matrix operand order repaired and checked by reading, not compiled, not run on hardware
//
#include <hip/hip_runtime.h>

typedef __attribute__((ext_vector_type(16))) _Float16 v16h;
typedef __attribute__((ext_vector_type(8)))  _Float16 h8;
typedef __attribute__((ext_vector_type(8)))  float    v8f;

#define ACT_NONE  0
#define ACT_RELU  1
#define ACT_PRELU 2

// ---------------------------------------------------------------------------
// Implicit-GEMM conv / transposed-conv via v_wmma_f32_16x16x32_f16.
//   M = Cout, N = Hout*Wout, K = Cin*KSZ*KSZ.
//   Block = 256 threads (8 waves): one 16-row M tile x 8 16-pixel N tiles.
//   LDS: weight slice (16 x Kpad f16, A-fragment striping) + packed tap
//   table (K -> {ci*Hin*Win [+ky*Win+kx], ky, kx}) built once per block.
//   Hot loop: 2x ds_load_b128 (A) + 16 branchless clamped gathers (B) + wmma.
// ---------------------------------------------------------------------------
template <int KSZ, int STRIDE, int PAD, int TR>
__global__ __launch_bounds__(256) void cen_conv_wmma(
    const float* __restrict__ in, const float* __restrict__ w,
    const float* __restrict__ bias, const float* __restrict__ res,
    float* __restrict__ out, int Cin, int Cout, int Hin, int Win,
    int Hout, int Wout, int act) {
  constexpr int KK2 = KSZ * KSZ;
  const int Ktot = Cin * KK2;
  const int Kpad = (Ktot + 31) & ~31;
  const int tid = threadIdx.x;
  const int mtile = blockIdx.y;

  __shared__ _Float16 sw[16 * 1152];   // 36 KB max
  __shared__ unsigned stap[1152];      // packed taps (unused for KSZ==1)
  // ---- cooperative weight staging ----------------------------------------
  for (int m = 0; m < 16; ++m) {
    int mg = mtile * 16 + m;
    for (int K = tid; K < Kpad; K += 256) {
      float wv = 0.f;
      if (mg < Cout && K < Ktot) {
        int ci = K / KK2;             // constant divisor -> mul/shift
        int r  = K - ci * KK2;
        int ky = r / KSZ, kx = r - (r / KSZ) * KSZ;
        if (TR) { ky = KSZ - 1 - ky; kx = KSZ - 1 - kx; }
        wv = w[((size_t)(mg * Cin + ci) * KK2) + ky * KSZ + kx];
      }
      sw[m * Kpad + K] = (_Float16)wv;
    }
  }
  // ---- tap table (only needed when KSZ > 1) ------------------------------
  if (KSZ > 1) {
    for (int K = tid; K < Kpad; K += 256) {
      unsigned u = 0;
      if (K < Ktot) {
        int ci = K / KK2;
        int r  = K - ci * KK2;
        int ky = r / KSZ, kx = r - (r / KSZ) * KSZ;
        int off = ci * Hin * Win + (TR ? 0 : ky * Win + kx);  // < 2^24
        u = (unsigned)off | ((unsigned)ky << 24) | ((unsigned)kx << 28);
      }
      stap[K] = u;
    }
  }
  __syncthreads();

  const int wave = tid >> 5, lane = tid & 31;
  const int ntile = blockIdx.x * 8 + wave;
  const int Npix = Hout * Wout;
  if (ntile * 16 >= Npix) return;     // wave-uniform; EXEC stays full below

  const int n = lane & 15;
  const int p = ntile * 16 + n;
  const int y = p / Wout;
  const int x = p - y * Wout;
  const bool hi = lane >= 16;
  const int ybase = y * STRIDE - PAD;
  const int xbase = x * STRIDE - PAD;
  const int base2 = ybase * Win + xbase;
  const bool pok = p < Npix;
  const _Float16* arow = sw + (lane & 15) * Kpad;
  __builtin_prefetch(in + p, 0, 1);   // global_prefetch_b8

  v8f acc = {};
  for (int kk0 = 0; kk0 < Ktot; kk0 += 32) {
    // ---- A fragment from LDS: two aligned 16B reads -----------------------
    const h8 a0 = *(const h8*)(arow + kk0 + (hi ? 8 : 0));
    const h8 a1 = *(const h8*)(arow + kk0 + 16 + (hi ? 8 : 0));
    v16h a;
#pragma unroll
    for (int e = 0; e < 8; ++e) { a[e] = a0[e]; a[8 + e] = a1[e]; }
    // ---- B fragment: branchless clamped im2col gather ---------------------
    v16h b;
#pragma unroll
    for (int e = 0; e < 16; ++e) {
      int K = kk0 + e + (e >= 8 ? 8 : 0) + (hi ? 8 : 0);
      bool ok; int adr;
      if (KSZ == 1) {                 // identity taps: addr = K*HW + p
        ok  = (K < Ktot) & pok;
        adr = K * (Hin * Win) + p;
      } else if (!TR) {
        unsigned u = stap[K];
        int ky = (int)((u >> 24) & 15), kx = (int)(u >> 28);
        int iy = ybase + ky, ix = xbase + kx;
        ok  = ((unsigned)iy < (unsigned)Hin) & ((unsigned)ix < (unsigned)Win) &
              (K < Ktot) & pok;
        adr = (int)(u & 0xFFFFFF) + base2;
      } else {                        // lhs-dilated (stride-2 deconv)
        constexpr int PE = KSZ - 1 - PAD;
        unsigned u = stap[K];
        int ky = (int)((u >> 24) & 15), kx = (int)(u >> 28);
        int dy = y - PE + ky, dx = x - PE + kx;
        int iy = dy >> 1, ix = dx >> 1;
        ok  = (dy >= 0) & (dx >= 0) & ((dy & 1) == 0) & ((dx & 1) == 0) &
              (iy < Hin) & (ix < Win) & (K < Ktot) & pok;
        adr = (int)(u & 0xFFFFFF) + iy * Win + ix;
      }
      int a2 = ok ? adr : 0;
      float t = in[a2];
      b[e] = (_Float16)(ok ? t : 0.f);
    }
    acc = __builtin_amdgcn_wmma_f32_16x16x32_f16(false, a, false, b,
                                                 (short)0, acc, false, false);
  }
  // ---- epilogue: bias + activation + residual -----------------------------
#pragma unroll
  for (int v = 0; v < 8; ++v) {
    int m  = v + (hi ? 8 : 0);
    int co = mtile * 16 + m;
    if (co < Cout && pok) {
      float val = acc[v] + bias[co];
      if (act == ACT_RELU)       val = fmaxf(val, 0.f);
      else if (act == ACT_PRELU) val = val >= 0.f ? val : 0.25f * val;
      size_t oi = (size_t)co * Npix + p;
      if (res) val += res[oi];
      out[oi] = val;
    }
  }
}

// ---------------------------------------------------------------------------
// Fused JBF kernel-generation + bilateral apply.
//   new[inner, p] = sum_g (W2[g*32+inner,:] . t[:,p] + b2[g*32+inner]) * patch
//   B fragment loaded once, reused across 49 back-to-back WMMAs per wave.
//   LDS: 49x16 weight rows (f16) + bias slice + packed tap table
//   (ch -> {cc*HH + dy*H + dx + 4096, dy+3, dx+3}); register accumulation,
//   deterministic, bi tensor never materialized.
// ---------------------------------------------------------------------------
__global__ __launch_bounds__(256) void cen_jbf_kg_agg(
    const float* __restrict__ t, const float* __restrict__ w,
    const float* __restrict__ bias, const float* __restrict__ img,
    float* __restrict__ nw, int H) {
  const int HH = H * H;
  const int tid = threadIdx.x;
  const int mtile = blockIdx.y;               // 0 or 1
  __shared__ _Float16 sw[49 * 16 * 32];       // 50176 B
  __shared__ float    sb[49 * 16];
  __shared__ unsigned spk[49 * 16];           // packed taps per (g, m)
  for (int idx = tid; idx < 49 * 16 * 32; idx += 256) {
    int g = idx >> 9, rem = idx & 511;
    int m = rem >> 5, K = rem & 31;
    sw[idx] = (_Float16)w[(size_t)(g * 32 + mtile * 16 + m) * 32 + K];
  }
  for (int idx = tid; idx < 49 * 16; idx += 256) {
    int g = idx >> 4, m = idx & 15;
    int ch = g * 32 + mtile * 16 + m;
    sb[idx] = bias[ch];
    int cc = ch / 49;                         // once per block, not per pixel
    int tt = ch - cc * 49;
    int dy = tt / 7 - 3, dx = (tt % 7) - 3;
    int off = cc * HH + dy * H + dx + 4096;   // >= 0, < 2^24
    spk[idx] = (unsigned)off | ((unsigned)(dy + 3) << 24) |
               ((unsigned)(dx + 3) << 28);
  }
  __syncthreads();

  const int wave = tid >> 5, lane = tid & 31;
  const int ntile = blockIdx.x * 8 + wave;
  if (ntile * 16 >= HH) return;               // wave-uniform
  const int n = lane & 15;
  const int p = ntile * 16 + n;
  const int y = p / H, x = p - y * H;
  const bool hi = lane >= 16;

  // B fragment loaded once: K = ci, coalesced across lanes
  v16h b;
#pragma unroll
  for (int e = 0; e < 16; ++e) {
    int K = e + (e >= 8 ? 8 : 0) + (hi ? 8 : 0);
    b[e] = (_Float16)t[(size_t)K * HH + p];
  }

  float accv[8] = {0.f, 0.f, 0.f, 0.f, 0.f, 0.f, 0.f, 0.f};
  for (int g = 0; g < 49; ++g) {
    const _Float16* arow = sw + ((size_t)(g * 16 + (lane & 15))) * 32;
    const h8 a0 = *(const h8*)(arow + (hi ? 8 : 0));
    const h8 a1 = *(const h8*)(arow + 16 + (hi ? 8 : 0));
    v16h a;
#pragma unroll
    for (int e = 0; e < 8; ++e) { a[e] = a0[e]; a[8 + e] = a1[e]; }
    v8f c = {};
    c = __builtin_amdgcn_wmma_f32_16x16x32_f16(false, a, false, b,
                                               (short)0, c, false, false);
#pragma unroll
    for (int v = 0; v < 8; ++v) {
      int m = v + (hi ? 8 : 0);
      unsigned u = spk[g * 16 + m];
      int dy = (int)((u >> 24) & 15) - 3;
      int dx = (int)(u >> 28) - 3;
      bool ok = ((unsigned)(y + dy) < (unsigned)H) &
                ((unsigned)(x + dx) < (unsigned)H);
      int adr = (int)(u & 0xFFFFFF) - 4096 + p;
      int a2 = ok ? adr : 0;
      float pv = img[a2];
      pv = ok ? pv : 0.f;
      accv[v] += (c[v] + sb[g * 16 + m]) * pv;
    }
  }
#pragma unroll
  for (int v = 0; v < 8; ++v) {
    int co = mtile * 16 + v + (hi ? 8 : 0);   // == ch % 32, g-invariant
    nw[(size_t)co * HH + p] = accv[v];
  }
}

// ---------------------------------------------------------------------------
// Elementwise helpers
// ---------------------------------------------------------------------------
__global__ void cen_sub(const float* a, const float* b, float* o, int n) {
  int i = blockIdx.x * blockDim.x + threadIdx.x;
  if (i < n) o[i] = a[i] - b[i];
}

__global__ void cen_sub_mean(const float* in, float* o, int HW) {
  int i = blockIdx.x * blockDim.x + threadIdx.x;
  if (i >= 3 * HW) return;
  const float mean[3] = {0.4488f, 0.4371f, 0.404f};
  o[i] = in[i] - mean[i / HW];
}

__global__ void cen_add_mean(float* o, int HW) {
  int i = blockIdx.x * blockDim.x + threadIdx.x;
  if (i >= 3 * HW) return;
  const float mean[3] = {0.4488f, 0.4371f, 0.404f};
  o[i] += mean[i / HW];
}

// pixel_shuffle r=2: in [4C,H,W] -> out [C,2H,2W]
__global__ void cen_pixel_shuffle2(const float* in, float* out, int C, int H, int W) {
  int i = blockIdx.x * blockDim.x + threadIdx.x;
  int n = C * 4 * H * W;
  if (i >= n) return;
  int W2 = 2 * W;
  int c   = i / (4 * H * W);
  int rem = i - c * 4 * H * W;
  int oy  = rem / W2;
  int ox  = rem - oy * W2;
  int r1 = oy & 1, r2 = ox & 1;
  out[i] = in[((size_t)(c * 4 + r1 * 2 + r2) * H + (oy >> 1)) * W + (ox >> 1)];
}

// ---------------------------------------------------------------------------
// Host orchestration
// ---------------------------------------------------------------------------
struct CP { const float* b; const float* w; };
struct RBp { CP c1, c2; };
struct KGp { CP c[5]; };
struct JBFp { CP jc1, jc2; KGp kg; CP tensor; };
struct EGEMp { JBFp jbf; RBp rb1, rb2; CP tensor; };
struct UPp { CP c1, d1, d2; };
struct DOWNp { CP c1, c2, d1; };

static inline void CONV(hipStream_t st, const float* in, CP p, const float* res,
                        float* out, int Cin, int Cout, int Hin, int Win,
                        int Hout, int Wout, int k, int stride, int pad,
                        int transposed, int act) {
  dim3 grid((Hout * Wout + 127) / 128, (Cout + 15) / 16, 1);
  dim3 blk(256, 1, 1);
  if (k == 3)
    cen_conv_wmma<3, 1, 1, 0><<<grid, blk, 0, st>>>(in, p.w, p.b, res, out,
        Cin, Cout, Hin, Win, Hout, Wout, act);
  else if (k == 1)
    cen_conv_wmma<1, 1, 0, 0><<<grid, blk, 0, st>>>(in, p.w, p.b, res, out,
        Cin, Cout, Hin, Win, Hout, Wout, act);
  else if (transposed)
    cen_conv_wmma<6, 2, 2, 1><<<grid, blk, 0, st>>>(in, p.w, p.b, res, out,
        Cin, Cout, Hin, Win, Hout, Wout, act);
  else
    cen_conv_wmma<6, 2, 2, 0><<<grid, blk, 0, st>>>(in, p.w, p.b, res, out,
        Cin, Cout, Hin, Win, Hout, Wout, act);
  (void)stride; (void)pad;
}

static void run_jbf(hipStream_t st, float* img /*in-place*/, const JBFp& P,
                    int H, float* cat2 /*guidance already at cat2+S*/,
                    float* ta, float* tb, float* nw) {
  size_t S = 32 * (size_t)H * H;
  hipMemcpyAsync(cat2, img, S * sizeof(float), hipMemcpyDeviceToDevice, st);
  CONV(st, cat2, P.tensor, nullptr, ta, 64, 32, H, H, H, H, 1, 1, 0, 0, ACT_PRELU);
  CONV(st, ta, P.kg.c[0], nullptr, tb, 32, 32, H, H, H, H, 3, 1, 1, 0, ACT_PRELU);
  CONV(st, tb, P.kg.c[1], nullptr, ta, 32, 32, H, H, H, H, 3, 1, 1, 0, ACT_PRELU);
  CONV(st, ta, P.kg.c[2], nullptr, tb, 32, 32, H, H, H, H, 3, 1, 1, 0, ACT_PRELU);
  CONV(st, tb, P.kg.c[3], nullptr, ta, 32, 32, H, H, H, H, 3, 1, 1, 0, ACT_PRELU);
  // fused KG 1x1 (1568 ch) + unfold*bi reduction, single launch
  dim3 grid((H * H + 127) / 128, 2, 1);
  cen_jbf_kg_agg<<<grid, 256, 0, st>>>(ta, P.kg.c[4].w, P.kg.c[4].b, img, nw, H);
  CONV(st, nw, P.jc1, nullptr, tb, 32, 32, H, H, H, H, 3, 1, 1, 0, ACT_PRELU);
  CONV(st, tb, P.jc2, img, img, 32, 32, H, H, H, H, 3, 1, 1, 0, ACT_NONE);
}

extern "C" void kernel_launch(void* const* d_in, const int* in_sizes, int n_in,
                              void* d_out, int out_size, void* d_ws, size_t ws_size,
                              hipStream_t stream) {
  (void)in_sizes; (void)n_in; (void)out_size; (void)ws_size;
  int ii = 0;
  const float* im_in = (const float*)d_in[ii++];
  const float* ed_in = (const float*)d_in[ii++];
  auto cp = [&]() { CP c; c.b = (const float*)d_in[ii++];
                    c.w = (const float*)d_in[ii++]; return c; };
  // params in jax-pytree order: dict keys sorted alphabetically, lists in order
  DOWNp downs[4];
  for (int i = 0; i < 4; ++i) { downs[i].c1 = cp(); downs[i].c2 = cp(); downs[i].d1 = cp(); }
  CP edge_recon = cp();
  CP edge_up    = cp();
  EGEMp eg[4];
  for (int i = 0; i < 4; ++i) {
    eg[i].jbf.jc1 = cp(); eg[i].jbf.jc2 = cp();
    for (int j = 0; j < 5; ++j) eg[i].jbf.kg.c[j] = cp();
    eg[i].jbf.tensor = cp();
    eg[i].rb1.c1 = cp(); eg[i].rb1.c2 = cp();
    eg[i].rb2.c1 = cp(); eg[i].rb2.c2 = cp();
    eg[i].tensor = cp();
  }
  CP eh1 = cp(), eh2 = cp();
  CP fusion = cp();
  CP ih1 = cp(), ih2 = cp();
  CP image_recon = cp();
  UPp image_up; image_up.c1 = cp(); image_up.d1 = cp(); image_up.d2 = cp();
  JBFp jbfF;
  jbfF.jc1 = cp(); jbfF.jc2 = cp();
  for (int j = 0; j < 5; ++j) jbfF.kg.c[j] = cp();
  jbfF.tensor = cp();
  RBp rb; rb.c1 = cp(); rb.c2 = cp();
  UPp ups[4];
  for (int i = 0; i < 4; ++i) { ups[i].c1 = cp(); ups[i].d1 = cp(); ups[i].d2 = cp(); }

  // ---- workspace layout (floats), ~80 MB total ---------------------------
  const int H = 96, HW = H * H, H2 = 192, HW2 = H2 * H2;
  const size_t S96 = 32 * (size_t)HW, S192 = 32 * (size_t)HW2;
  float* base = (float*)d_ws; size_t off = 0;
  auto alloc = [&](size_t n) { float* p = base + off; off += n; return p; };
  float* T3    = alloc(3 * HW);
  float* T128  = alloc(128 * (size_t)HW);
  float* IMF   = alloc(S96);
  float* EDGE  = alloc(S96);
  float* DCAT  = alloc(4 * S96);
  float* UCAT  = alloc(5 * S192);
  float* UH0   = alloc(S192);
  float* T96a  = alloc(S96);
  float* T96b  = alloc(S96);
  float* NEW96 = alloc(S96);
  float* CAT64 = alloc(2 * S96);
  float* T192a = alloc(S192);
  float* FUS   = alloc(S192);
  float* EUP   = alloc(S192);
  float* CAT192= alloc(2 * S192);
  float* T192b = alloc(S192);
  float* T192c = alloc(S192);
  float* NEW192= alloc(S192);

  hipStream_t st = stream;
  // ---- heads --------------------------------------------------------------
  cen_sub_mean<<<(3 * HW + 255) / 256, 256, 0, st>>>(im_in, T3, HW);
  CONV(st, T3,   ih1, nullptr, T128, 3, 128, H, H, H, H, 3, 1, 1, 0, ACT_PRELU);
  CONV(st, T128, ih2, nullptr, IMF, 128, 32, H, H, H, H, 3, 1, 1, 0, ACT_PRELU);
  CONV(st, ed_in, eh1, nullptr, T128, 3, 128, H, H, H, H, 3, 1, 1, 0, ACT_PRELU);
  CONV(st, T128, eh2, nullptr, EDGE, 128, 32, H, H, H, H, 3, 1, 1, 0, ACT_PRELU);
  CONV(st, EDGE, rb.c1, nullptr, T96b, 32, 32, H, H, H, H, 3, 1, 1, 0, ACT_RELU);
  CONV(st, T96b, rb.c2, EDGE, EDGE, 32, 32, H, H, H, H, 3, 1, 1, 0, ACT_NONE);

  // ---- DBPN groups + edge-guided modules ----------------------------------
  const float* down_ten = IMF;
  for (int i = 0; i < 4; ++i) {
    // upblock: h0 = prelu(deconv(x)); l0 = prelu(conv(h0)); h1 = prelu(deconv(l0-x)) + h0
    CONV(st, down_ten, ups[i].d1, nullptr, UH0, 32, 32, H, H, H2, H2, 6, 2, 2, 1, ACT_PRELU);
    CONV(st, UH0, ups[i].c1, nullptr, T96a, 32, 32, H2, H2, H, H, 6, 2, 2, 0, ACT_PRELU);
    cen_sub<<<((int)S96 + 255) / 256, 256, 0, st>>>(T96a, down_ten, T96b, (int)S96);
    float* upslot = UCAT + (size_t)i * S192;
    CONV(st, T96b, ups[i].d2, UH0, upslot, 32, 32, H, H, H2, H2, 6, 2, 2, 1, ACT_PRELU);
    // downblock: l0 = prelu(conv(u)); h0 = prelu(deconv(l0)); l1 = prelu(conv(h0-u)) + l0
    CONV(st, upslot, downs[i].c1, nullptr, T96a, 32, 32, H2, H2, H, H, 6, 2, 2, 0, ACT_PRELU);
    CONV(st, T96a, downs[i].d1, nullptr, UH0, 32, 32, H, H, H2, H2, 6, 2, 2, 1, ACT_PRELU);
    cen_sub<<<((int)S192 + 255) / 256, 256, 0, st>>>(UH0, upslot, T192a, (int)S192);
    float* dslot = DCAT + (size_t)i * S96;
    CONV(st, T192a, downs[i].c2, T96a, dslot, 32, 32, H2, H2, H, H, 6, 2, 2, 0, ACT_PRELU);
    down_ten = dslot;
    // feature = prelu(conv1x1(down_concat)) -> second half of CAT64
    CONV(st, DCAT, eg[i].tensor, nullptr, CAT64 + S96, 32 * (i + 1), 32,
         H, H, H, H, 1, 1, 0, 0, ACT_PRELU);
    // edge_ten = jbf(edge_ten, feature); then 2 resblocks
    run_jbf(st, EDGE, eg[i].jbf, H, CAT64, T96a, T96b, NEW96);
    CONV(st, EDGE, eg[i].rb1.c1, nullptr, T96b, 32, 32, H, H, H, H, 3, 1, 1, 0, ACT_RELU);
    CONV(st, T96b, eg[i].rb1.c2, EDGE, EDGE, 32, 32, H, H, H, H, 3, 1, 1, 0, ACT_NONE);
    CONV(st, EDGE, eg[i].rb2.c1, nullptr, T96b, 32, 32, H, H, H, H, 3, 1, 1, 0, ACT_RELU);
    CONV(st, T96b, eg[i].rb2.c2, EDGE, EDGE, 32, 32, H, H, H, H, 3, 1, 1, 0, ACT_NONE);
  }

  // ---- tail: final upblock, fusion, edge upsample, final JBF, recon -------
  CONV(st, down_ten, image_up.d1, nullptr, UH0, 32, 32, H, H, H2, H2, 6, 2, 2, 1, ACT_PRELU);
  CONV(st, UH0, image_up.c1, nullptr, T96a, 32, 32, H2, H2, H, H, 6, 2, 2, 0, ACT_PRELU);
  cen_sub<<<((int)S96 + 255) / 256, 256, 0, st>>>(T96a, down_ten, T96b, (int)S96);
  CONV(st, T96b, image_up.d2, UH0, UCAT + 4 * S192, 32, 32, H, H, H2, H2, 6, 2, 2, 1, ACT_PRELU);
  CONV(st, UCAT, fusion, nullptr, FUS, 160, 32, H2, H2, H2, H2, 1, 1, 0, 0, ACT_NONE);
  CONV(st, EDGE, edge_up, nullptr, T128, 32, 128, H, H, H, H, 3, 1, 1, 0, ACT_NONE);
  cen_pixel_shuffle2<<<((int)S192 + 255) / 256, 256, 0, st>>>(T128, EUP, 32, H, H);
  hipMemcpyAsync(CAT192 + S192, EUP, S192 * sizeof(float),
                 hipMemcpyDeviceToDevice, st);
  run_jbf(st, FUS, jbfF, H2, CAT192, T192b, T192c, NEW192);
  float* out = (float*)d_out;
  CONV(st, FUS, image_recon, nullptr, out, 32, 3, H2, H2, H2, H2, 3, 1, 1, 0, ACT_NONE);
  cen_add_mean<<<(3 * HW2 + 255) / 256, 256, 0, st>>>(out, HW2);
  CONV(st, EUP, edge_recon, nullptr, out + 3 * (size_t)HW2, 32, 1,
       H2, H2, H2, H2, 3, 1, 1, 0, ACT_NONE);
}
